// Net_85100482003585
// MI455X (gfx1250) — compile-verified, matrix operand order checked by reading, not run on hardware
//
#include <hip/hip_runtime.h>

typedef __attribute__((ext_vector_type(16))) _Float16 v16h;
typedef __attribute__((ext_vector_type(8)))  _Float16 v8h;
typedef __attribute__((ext_vector_type(8)))  float    v8f;
typedef __attribute__((ext_vector_type(4)))  float    v4f;

// K index within a 32-wide K slab for 16-bit A/B WMMA fragments (wave32).
// ISA 7.12.2: lanes 0-15: elems 0-7 -> K0-7, elems 8-15 -> K16-23;
//             lanes16-31: elems 0-7 -> K8-15, elems 8-15 -> K24-31.
// => per lane: two contiguous 8-element runs at kbase and kbase+16,
//    with kbase = (lane>>4)*8.
__device__ __forceinline__ int frag_k(int lane, int e) {
  return ((e >> 3) << 4) + ((lane >> 4) << 3) + (e & 7);
}

union Frag16 { v16h v; v8h h[2]; };

// Fragment from LDS: two 16-byte (8 x f16) contiguous chunks -> 2x ds_load_b128.
__device__ __forceinline__ v16h load_frag_lds(const _Float16* p) {
  Frag16 f;
  f.h[0] = *(const v8h*)(p);
  f.h[1] = *(const v8h*)(p + 16);
  return f.v;
}

// Fragment from global f32: two 8-float contiguous runs (4x b128) + pack to f16.
__device__ __forceinline__ v16h load_frag_f32(const float* p) {
  Frag16 f;
  #pragma unroll
  for (int c = 0; c < 2; ++c) {
    v4f x0 = *(const v4f*)(p + c * 16);
    v4f x1 = *(const v4f*)(p + c * 16 + 4);
    v8h h;
    #pragma unroll
    for (int e = 0; e < 4; ++e) {
      h[e]     = (_Float16)x0[e];
      h[e + 4] = (_Float16)x1[e];
    }
    f.h[c] = h;
  }
  return f.v;
}

// Fused embed: conv1(relu) -> conv2(relu, WMMA) -> fc1(relu, WMMA) -> fc2.
// One block = 16 samples, 128 threads = 4 wave32s.
__global__ __launch_bounds__(128) void embed_kernel(
    const float* __restrict__ X,     // (M,1,8,8)
    const float* __restrict__ c1w,   // (32,1,3,3)
    const float* __restrict__ c1b,   // (32)
    const float* __restrict__ c2w,   // (64,32,3,3)
    const float* __restrict__ c2b,   // (64)
    const float* __restrict__ fc1w,  // (128,1024)
    const float* __restrict__ fc2w,  // (10,128)
    float* __restrict__ out,         // (M,10)
    int M)
{
  __shared__ __align__(16) char smem[55296];
  _Float16* a2  = (_Float16*)smem;                  // [16][1024] fc1 input (f16, K-contig)
  _Float16* a1t = (_Float16*)(smem + 32768);        // [8][36 pix][32 ci] conv1 out, ch-innermost
  _Float16* a3  = (_Float16*)(smem + 32768);        // [16][128] fc1 out (aliases a1t)
  float*    xin = (float*)(smem + 32768 + 18432);   // [16][8][8] raw input

  const int tid  = threadIdx.x;
  const int lane = tid & 31;
  const int wv   = tid >> 5;
  const int krun = (lane >> 4) << 3;   // per-lane K-run base within a 32-slab
  const int s0   = blockIdx.x * 16;

  // ---- stage input tile ----
  for (int i = tid; i < 16 * 64; i += 128) {
    int s = i >> 6, pix = i & 63;
    int sg = s0 + s;
    xin[i] = (sg < M) ? X[sg * 64 + pix] : 0.f;
  }

  // Process samples in two halves of 8 so conv1 scratch stays small.
  for (int hh = 0; hh < 2; ++hh) {
    __syncthreads();
    // ---- conv1 + relu (fp32 VALU; ~2% of FLOPs), channel-innermost store ----
    for (int i = tid; i < 8 * 36 * 32; i += 128) {
      int s = i / 1152, r = i - s * 1152;
      int p = r >> 5,   c = r & 31;
      int py = p / 6,   px = p - py * 6;
      const float* xs = xin + (hh * 8 + s) * 64 + py * 8 + px;
      const float* wc = c1w + c * 9;
      float acc = c1b[c];
      #pragma unroll
      for (int ky = 0; ky < 3; ++ky)
        #pragma unroll
        for (int kx = 0; kx < 3; ++kx)
          acc += xs[ky * 8 + kx] * wc[ky * 3 + kx];
      a1t[i] = (_Float16)fmaxf(acc, 0.f);
    }
    __syncthreads();

    // ---- conv2 as implicit-GEMM WMMA:
    //      rows = 8 samples x 16 positions = 128, cols = 64 ch, K = 288.
    //      K permuted as k = kk*32 + ci (kernel-position-major) so A-fragment
    //      K-runs are contiguous input channels -> ds_load_b128. ----
    {
      const int ct = wv;                // each wave owns one 16-column tile
      v8f acc[8] = {};
      for (int ks = 0; ks < 9; ++ks) {  // slab = kernel position kk
        const int ky = ks / 3, kx = ks - ky * 3;   // uniform scalars
        // B fragment: c2w[n][ci*9 + kk], strided gather; hoisted across 8 rt.
        v16h bf;
        int n = ct * 16 + (lane & 15);
        #pragma unroll
        for (int e = 0; e < 16; ++e) {
          int ci = frag_k(lane, e);
          bf[e] = (_Float16)c2w[n * 288 + ci * 9 + ks];
        }
        #pragma unroll
        for (int rt = 0; rt < 8; ++rt) {
          int m = rt * 16 + (lane & 15);
          int s = m >> 4, p = m & 15;
          int pix = ((p >> 2) + ky) * 6 + (p & 3) + kx;
          v16h af = load_frag_lds(a1t + s * 1152 + pix * 32 + krun);
          acc[rt] = __builtin_amdgcn_wmma_f32_16x16x32_f16(
              false, af, false, bf, (short)0, acc[rt], false, false);
        }
      }
      // bias + relu epilogue; store flattened NCHW (ch*16 + pos) as f16
      #pragma unroll
      for (int rt = 0; rt < 8; ++rt) {
        int n = ct * 16 + (lane & 15);
        float bias = c2b[n];
        #pragma unroll
        for (int r = 0; r < 8; ++r) {
          int m = rt * 16 + ((lane >> 4) * 8 + r);
          int s = m >> 4, p = m & 15;
          float v = fmaxf(acc[rt][r] + bias, 0.f);
          a2[(hh * 8 + s) * 1024 + n * 16 + p] = (_Float16)v;
        }
      }
    }
  }
  __syncthreads();

  // ---- fc1 WMMA: 16 x 128, K = 1024; relu.
  //      A rows are K-contiguous -> ds_load_b128; B via global b128 + cvt. ----
  {
    v8f acc[2] = {};
    for (int ks = 0; ks < 32; ++ks) {
      v16h af = load_frag_lds(a2 + (lane & 15) * 1024 + ks * 32 + krun);
      #pragma unroll
      for (int c = 0; c < 2; ++c) {
        int n = (wv * 2 + c) * 16 + (lane & 15);
        v16h bf = load_frag_f32(fc1w + n * 1024 + ks * 32 + krun);
        acc[c] = __builtin_amdgcn_wmma_f32_16x16x32_f16(
            false, af, false, bf, (short)0, acc[c], false, false);
      }
    }
    #pragma unroll
    for (int c = 0; c < 2; ++c) {
      int n = (wv * 2 + c) * 16 + (lane & 15);
      #pragma unroll
      for (int r = 0; r < 8; ++r) {
        int m = (lane >> 4) * 8 + r;
        a3[m * 128 + n] = (_Float16)fmaxf(acc[c][r], 0.f);
      }
    }
  }
  __syncthreads();

  // ---- fc2: 16 x 10, K = 128 (tiny; fp32 VALU) ----
  for (int t = tid; t < 160; t += 128) {
    int s = t / 10, j = t - s * 10;
    float acc = 0.f;
    for (int k = 0; k < 128; ++k)
      acc += (float)a3[s * 128 + k] * fc2w[j * 128 + k];
    int sg = s0 + s;
    if (sg < M) out[sg * 10 + j] = acc;
  }
}

// Gaussian-KDE regression: one block per query row b, reduce over N train rows.
__global__ __launch_bounds__(256) void kde_kernel(
    const float* __restrict__ Xw,   // (N,10)
    const float* __restrict__ Zw,   // (B,10)
    const float* __restrict__ Y,    // (N,10)
    const float* __restrict__ hptr, // scalar
    float* __restrict__ out,        // (B,10)
    int N)
{
  __shared__ float s_num[10];
  __shared__ float s_den[10];
  const int b = blockIdx.x, tid = threadIdx.x;
  if (tid < 10) { s_num[tid] = 0.f; s_den[tid] = 0.f; }
  const float invh = 1.0f / hptr[0];
  float z[10];
  #pragma unroll
  for (int d = 0; d < 10; ++d) z[d] = Zw[b * 10 + d];
  __syncthreads();

  float num[10] = {}, den[10] = {};
  for (int n = tid; n < N; n += 256) {
    #pragma unroll
    for (int d = 0; d < 10; ++d) {
      float t = (Xw[n * 10 + d] - z[d]) * invh;
      float e = __expf(-0.5f * t * t);
      den[d] += e;
      num[d] += e * Y[n * 10 + d];
    }
  }
  // wave32 shuffle reduction, then LDS float atomics across the 8 waves
  #pragma unroll
  for (int off = 16; off > 0; off >>= 1) {
    #pragma unroll
    for (int d = 0; d < 10; ++d) {
      num[d] += __shfl_down(num[d], off, 32);
      den[d] += __shfl_down(den[d], off, 32);
    }
  }
  if ((tid & 31) == 0) {
    for (int d = 0; d < 10; ++d) {
      atomicAdd(&s_num[d], num[d]);
      atomicAdd(&s_den[d], den[d]);
    }
  }
  __syncthreads();
  if (tid < 10) out[b * 10 + tid] = s_num[tid] / s_den[tid];
}

extern "C" void kernel_launch(void* const* d_in, const int* in_sizes, int n_in,
                              void* d_out, int out_size, void* d_ws, size_t ws_size,
                              hipStream_t stream) {
  const float* x    = (const float*)d_in[0];  // (B,1,8,8)
  const float* trX  = (const float*)d_in[1];  // (N,1,8,8)
  const float* Y    = (const float*)d_in[2];  // (N,10)
  const float* c1w  = (const float*)d_in[3];
  const float* c1b  = (const float*)d_in[4];
  const float* c2w  = (const float*)d_in[5];
  const float* c2b  = (const float*)d_in[6];
  const float* fc1w = (const float*)d_in[7];
  const float* fc2w = (const float*)d_in[8];
  const float* h    = (const float*)d_in[9];

  const int B = in_sizes[0] / 64;
  const int N = in_sizes[1] / 64;

  float* Zw = (float*)d_ws;       // B*10 floats
  float* Xw = Zw + 4096;          // N*10 floats (16 KB aligned offset)

  embed_kernel<<<(N + 15) / 16, 128, 0, stream>>>(trX, c1w, c1b, c2w, c2b,
                                                  fc1w, fc2w, Xw, N);
  embed_kernel<<<(B + 15) / 16, 128, 0, stream>>>(x, c1w, c1b, c2w, c2b,
                                                  fc1w, fc2w, Zw, B);
  kde_kernel<<<B, 256, 0, stream>>>(Xw, Zw, Y, h, (float*)d_out, N);
}
